// LinearScaSplitNN_40570261078345
// MI455X (gfx1250) — matrix operand order repaired and checked by reading, not compiled, not run on hardware
//
#include <hip/hip_runtime.h>

// ---------------------------------------------------------------------------
// MI455X (gfx1250) LCA network. All heavy math on v_wmma_f32_16x16x32_f16.
// LCA tap loop: dy-outer (no unroll -> single acc phi, in-place WMMA),
// dx fully unrolled so B/A addresses fold into ds/global immediate offsets.
// MLP chain: f16 zero-padded operands -> branch-free vector-load GEMM.
// ---------------------------------------------------------------------------

typedef __attribute__((ext_vector_type(16))) _Float16 v16h;
typedef __attribute__((ext_vector_type(8)))  _Float16 v8h;
typedef __attribute__((ext_vector_type(8)))  float    v8f;

#define LCA_LAMBDA 0.5f
#define LCA_ETA    0.01f     // 1 / TAU
#define HW         28
#define LW         38        // LDS tile width (36 needed + 2 pad)

__device__ __forceinline__ v8f wmma_f16(v16h a, v16h b, v8f c) {
  return __builtin_amdgcn_wmma_f32_16x16x32_f16(
      false, a, false, b, (short)0, c, false, false);
}

__device__ __forceinline__ v16h cat8(v8h lo, v8h hi) {
  return __builtin_shufflevector(lo, hi, 0, 1, 2, 3, 4, 5, 6, 7,
                                         8, 9, 10, 11, 12, 13, 14, 15);
}

// ---------------------------------------------------------------------------
__global__ void k_normalize(const float* __restrict__ W, float* __restrict__ Wn,
                            int nf, int fsz) {
  int f = blockIdx.x * blockDim.x + threadIdx.x;
  if (f >= nf) return;
  const float* w = W + f * fsz;
  float s = 0.f;
  for (int i = 0; i < fsz; ++i) s += w[i] * w[i];
  float inv = rsqrtf(s + 1e-12f);
  float* o = Wn + f * fsz;
  for (int i = 0; i < fsz; ++i) o[i] = w[i] * inv;
}

// G[f,g,p,q] = sum_{c,a,b} Wn[f,c,p+a-4,q+b-4]*Wn[g,c,a,b], stored as f16
// tap matrices Gt[tap][32f][32g] (zero-padded) ready for WMMA A-operand rows.
__global__ void k_gram(const float* __restrict__ Wn, _Float16* __restrict__ Gt,
                       int F, int C) {
  int idx = blockIdx.x * blockDim.x + threadIdx.x;
  if (idx >= 81 * 32 * 32) return;
  int tap = idx >> 10;
  int f = (idx >> 5) & 31;
  int g = idx & 31;
  float s = 0.f;
  if (f < F && g < F) {
    int p = tap / 9, q = tap - (tap / 9) * 9;
    for (int c = 0; c < C; ++c)
      for (int a = 0; a < 5; ++a) {
        int ia = p + a - 4;
        if (ia < 0 || ia >= 5) continue;
        for (int bb = 0; bb < 5; ++bb) {
          int jb = q + bb - 4;
          if (jb < 0 || jb >= 5) continue;
          s += Wn[((f * C + c) * 5 + ia) * 5 + jb] *
               Wn[((g * C + c) * 5 + a) * 5 + bb];
        }
      }
  }
  Gt[idx] = (_Float16)s;
}

__global__ void k_drive(const float* __restrict__ in, const float* __restrict__ Wn,
                        float* __restrict__ out, int B, int C, int F) {
  int idx = blockIdx.x * blockDim.x + threadIdx.x;
  if (idx >= B * F * HW * HW) return;
  int x = idx % HW;
  int y = (idx / HW) % HW;
  int f = (idx / (HW * HW)) % F;
  int b = idx / (HW * HW * F);
  float s = 0.f;
  for (int c = 0; c < C; ++c)
    for (int i = 0; i < 5; ++i) {
      int sy = y + i - 2;
      if (sy < 0 || sy >= HW) continue;
      for (int j = 0; j < 5; ++j) {
        int sx = x + j - 2;
        if (sx < 0 || sx >= HW) continue;
        s += in[((b * C + c) * HW + sy) * HW + sx] *
             Wn[((f * C + c) * 5 + i) * 5 + j];
      }
    }
  out[idx] = s;
}

__global__ void k_zero(float* __restrict__ p, int n) {
  int i = blockIdx.x * blockDim.x + threadIdx.x;
  if (i < n) p[i] = 0.f;
}

__global__ void k_thresh(const float* __restrict__ u, float* __restrict__ a, int n) {
  int i = blockIdx.x * blockDim.x + threadIdx.x;
  if (i < n) a[i] = fmaxf(u[i] - LCA_LAMBDA, 0.f);
}

// relu(u - lambda) into f16 rows padded from 28 -> 32 columns (GEMM K-pad).
__global__ void k_thresh_pad(const float* __restrict__ u, _Float16* __restrict__ o,
                             int rows) {
  int idx = blockIdx.x * blockDim.x + threadIdx.x;
  if (idx >= rows * 32) return;
  int row = idx >> 5, xc = idx & 31;
  float v = 0.f;
  if (xc < HW) v = fmaxf(u[row * HW + xc] - LCA_LAMBDA, 0.f);
  o[idx] = (_Float16)v;
}

// f32 [N,K] weights -> f16 [Npad,Kpad] zero-padded.
__global__ void k_cvt_pad(const float* __restrict__ W, _Float16* __restrict__ Wp,
                          int N, int K, int Npad, int Kpad) {
  int idx = blockIdx.x * blockDim.x + threadIdx.x;
  if (idx >= Npad * Kpad) return;
  int rw = idx / Kpad, k = idx - rw * Kpad;
  float v = (rw < N && k < K) ? W[rw * K + k] : 0.f;
  Wp[idx] = (_Float16)v;
}

// ---------------------------------------------------------------------------
// One LCA iteration (templated on feature count F).
// Grid: 8 images x 4 seven-row strips; 4 waves/WG, 4 live tiles per wave.
// LDS: sa[(ly*38+lx)*32 + g] f16 (15 x 38 x 32 = 36.5 KB, zero halo).
// ---------------------------------------------------------------------------
template <int F>
__global__ __launch_bounds__(128) void k_lca_step(
    const float* __restrict__ u_in, float* __restrict__ u_out,
    const float* __restrict__ drive, const _Float16* __restrict__ Gt) {
  constexpr bool TWO_MT = (F > 16);
  __shared__ __align__(32) _Float16 sa[15 * LW * 32];

  const int b = blockIdx.x >> 2;
  const int rg = blockIdx.x & 3;
  const int rowbase = 7 * rg;

  // Stage a = relu(u - lambda): coalesced global reads, interleaved LDS writes.
  for (int idx = threadIdx.x; idx < 32 * 15 * LW; idx += 128) {
    int g = idx / (15 * LW);
    int pos = idx - g * (15 * LW);
    int ly = pos / LW;
    int lx = pos - ly * LW;
    int sy = rowbase + ly - 4;
    int sx = lx - 4;
    float v = 0.f;
    if (g < F && (unsigned)sy < HW && (unsigned)sx < HW)
      v = fmaxf(u_in[((b * F + g) * HW + sy) * HW + sx] - LCA_LAMBDA, 0.f);
    sa[(pos << 5) + g] = (_Float16)v;
  }
  __syncthreads();

  const int wave = threadIdx.x >> 5;
  const int lane = threadIdx.x & 31;
  const int n = lane & 15;
  const int hi = lane >> 4;
  const int koff = hi << 4;

  // 4 half-row tiles per wave (tiles 14..15 are dead, clamped to tile 13).
  int yy[4], x0[4], live[4], base[4];
  #pragma unroll
  for (int i = 0; i < 4; ++i) {
    int t = wave * 4 + i;
    live[i] = (t < 14);
    int tc = t > 13 ? 13 : t;
    yy[i] = tc >> 1;             // row within strip, 0..6
    x0[i] = (tc & 1) * 14;       // half-row start column
    // LDS base (halves) at tap (dy=0,dx=0): center offset +4,+4
    base[i] = (((yy[i] + 4) * LW + x0[i] + n + 4) << 5) + koff;
  }

  v8f acc0[4] = {};
  v8f acc1[4] = {};

  // dy outer (not unrolled -> single phi per accumulator, in-place WMMA);
  // dx inner fully unrolled -> ds/global immediate-offset addressing.
#pragma clang loop unroll(disable)
  for (int dy = 0; dy < 9; ++dy) {
    __builtin_prefetch(Gt + (dy + 1) * 9 * 1024, 0, 0);
    const _Float16* gtap = Gt + dy * 9 * 1024 + (lane & 15) * 32;
    const int rowoff = (((dy - 4) * LW - 4) << 5);   // scalar, wave-uniform
    int addr0 = base[0] + rowoff;
    int addr1 = base[1] + rowoff;
    int addr2 = base[2] + rowoff;
    int addr3 = base[3] + rowoff;

#pragma unroll
    for (int dx = 0; dx < 9; ++dx) {
      const _Float16* ga = gtap + dx * 1024;
      const v16h af0 = cat8(*(const v8h*)(ga + hi * 16),
                            *(const v8h*)(ga + 8 + hi * 16));
      v16h af1 = {};
      if (TWO_MT) {
        const _Float16* gb = ga + 16 * 32;
        af1 = cat8(*(const v8h*)(gb + hi * 16),
                   *(const v8h*)(gb + 8 + hi * 16));
      }
      const int d = dx << 5;   // folds into ds_load immediate offsets
      const v16h bf0 = *(const v16h*)(sa + addr0 + d);
      const v16h bf1 = *(const v16h*)(sa + addr1 + d);
      const v16h bf2 = *(const v16h*)(sa + addr2 + d);
      const v16h bf3 = *(const v16h*)(sa + addr3 + d);
      acc0[0] = wmma_f16(af0, bf0, acc0[0]);
      acc0[1] = wmma_f16(af0, bf1, acc0[1]);
      acc0[2] = wmma_f16(af0, bf2, acc0[2]);
      acc0[3] = wmma_f16(af0, bf3, acc0[3]);
      if (TWO_MT) {
        acc1[0] = wmma_f16(af1, bf0, acc1[0]);
        acc1[1] = wmma_f16(af1, bf1, acc1[1]);
        acc1[2] = wmma_f16(af1, bf2, acc1[2]);
        acc1[3] = wmma_f16(af1, bf3, acc1[3]);
      }
    }
  }

  // u' = u + eta*(drive - u - inhib + a)
#pragma unroll
  for (int i = 0; i < 4; ++i) {
    if (!live[i]) continue;                // wave-uniform
    const int y = rowbase + yy[i];
    const int x = x0[i] + n;
    const bool ok = (n < 14);              // x < 28 implied
#pragma unroll
    for (int v = 0; v < 8; ++v) {
      int f = v + hi * 8;
      if (ok && f < F) {
        int idx = ((b * F + f) * HW + y) * HW + x;
        float uo = u_in[idx];
        float aa = fmaxf(uo - LCA_LAMBDA, 0.f);
        u_out[idx] = uo + LCA_ETA * (drive[idx] - uo - acc0[i][v] + aa);
      }
      if (TWO_MT) {
        int f2 = 16 + v + hi * 8;
        if (ok && f2 < F) {
          int idx = ((b * F + f2) * HW + y) * HW + x;
          float uo = u_in[idx];
          float aa = fmaxf(uo - LCA_LAMBDA, 0.f);
          u_out[idx] = uo + LCA_ETA * (drive[idx] - uo - acc1[i][v] + aa);
        }
      }
    }
  }
}

// ---------------------------------------------------------------------------
// f16 GEMM: C16[M,Npad] = act(A16[M,Kpad] * W16[Npad,Kpad]^T + bias[N]).
// Zero-padded operands -> no bounds checks in the K loop; all fragment
// loads are contiguous b128 vector loads.
// ---------------------------------------------------------------------------
template <int RELU>
__global__ __launch_bounds__(128) void k_gemm16(
    const _Float16* __restrict__ A, const _Float16* __restrict__ W,
    const float* __restrict__ bias, _Float16* __restrict__ C,
    int M, int Kpad, int N, int Npad) {
  const int wave = threadIdx.x >> 5;
  const int lane = threadIdx.x & 31;
  const int ntn = Npad >> 4;
  const int ntm = M >> 4;          // M is a multiple of 16
  const int tile = blockIdx.x * 4 + wave;
  if (tile >= ntm * ntn) return;   // wave-uniform
  const int mt = tile / ntn;
  const int nt = tile - mt * ntn;
  const int hi = lane >> 4;
  const int koff = hi << 4;

  const _Float16* pa = A + (size_t)(mt * 16 + (lane & 15)) * Kpad;
  const _Float16* pw = W + (size_t)(nt * 16 + (lane & 15)) * Kpad;

  v8f acc = {};
#pragma clang loop unroll(disable)
  for (int kk = 0; kk < Kpad; kk += 32) {
    const v16h af = cat8(*(const v8h*)(pa + kk + hi * 16),
                         *(const v8h*)(pa + kk + 8 + hi * 16));
    const v16h bf = *(const v16h*)(pw + kk + koff);
    acc = wmma_f16(af, bf, acc);
  }

  const int n = lane & 15;
  const int ccol = nt * 16 + n;
  const bool colok = (ccol < N);
  const float bv = colok ? bias[ccol] : 0.f;
#pragma unroll
  for (int v = 0; v < 8; ++v) {
    const int r = mt * 16 + v + hi * 8;
    float o = acc[v] + bv;
    if (RELU) o = fmaxf(o, 0.f);
    C[(size_t)r * Npad + ccol] = colok ? (_Float16)o : (_Float16)0.f;
  }
}

// ---------------------------------------------------------------------------
// out[8,10] = H16[8, 784x(512-pad-500)] . tp_w[10,392000]^T + tp_b.
// Memory-bound; division-free indexing over the padded f16 activations.
// ---------------------------------------------------------------------------
__global__ void k_final(const _Float16* __restrict__ H, const float* __restrict__ Wt,
                        const float* __restrict__ bias, float* __restrict__ out) {
  __shared__ float red[256];
  const int b = blockIdx.x / 10;
  const int o = blockIdx.x - b * 10;
  const float* w = Wt + (size_t)o * 392000;
  float s = 0.f;
  for (int r = 0; r < 784; ++r) {
    const _Float16* hr = H + (size_t)(b * 784 + r) * 512;
    const float* wr = w + r * 500;
    for (int c = threadIdx.x; c < 500; c += 256)
      s += (float)hr[c] * wr[c];
  }
  red[threadIdx.x] = s;
  __syncthreads();
  for (int st = 128; st > 0; st >>= 1) {
    if (threadIdx.x < st) red[threadIdx.x] += red[threadIdx.x + st];
    __syncthreads();
  }
  if (threadIdx.x == 0) out[b * 10 + o] = red[0] + bias[o];
}

// ---------------------------------------------------------------------------
extern "C" void kernel_launch(void* const* d_in, const int* in_sizes, int n_in,
                              void* d_out, int out_size, void* d_ws, size_t ws_size,
                              hipStream_t stream) {
  (void)in_sizes; (void)n_in; (void)out_size; (void)ws_size;
  const float* x     = (const float*)d_in[0];
  const float* W1    = (const float*)d_in[1];
  const float* W2    = (const float*)d_in[2];
  const float* fp_w  = (const float*)d_in[3];
  const float* fp_b  = (const float*)d_in[4];
  const float* sp_w1 = (const float*)d_in[5];
  const float* sp_b1 = (const float*)d_in[6];
  const float* sp_w2 = (const float*)d_in[7];
  const float* sp_b2 = (const float*)d_in[8];
  const float* sp_w3 = (const float*)d_in[9];
  const float* sp_b3 = (const float*)d_in[10];
  const float* tp_w  = (const float*)d_in[11];
  const float* tp_b  = (const float*)d_in[12];
  float* out = (float*)d_out;

  char* ws = (char*)d_ws;
  size_t off = 0;
  auto alloc = [&](size_t bytes) -> void* {
    void* p = (void*)(ws + off);
    off += (bytes + 255) & ~(size_t)255;
    return p;
  };

  const int N1 = 8 * 16 * HW * HW;   // 100352
  const int N2 = 8 * 28 * HW * HW;   // 175616
  const int M  = 8 * 28 * HW;        // 6272 MLP rows

  float*    Wn1    = (float*)alloc(16 * 25 * 4);
  float*    Wn2    = (float*)alloc(28 * 400 * 4);
  _Float16* Gt1    = (_Float16*)alloc(81 * 32 * 32 * 2);
  _Float16* Gt2    = (_Float16*)alloc(81 * 32 * 32 * 2);
  float*    drive1 = (float*)alloc((size_t)N1 * 4);
  float*    u1a    = (float*)alloc((size_t)N1 * 4);
  float*    u1b    = (float*)alloc((size_t)N1 * 4);
  float*    h1     = (float*)alloc((size_t)N1 * 4);
  float*    drive2 = (float*)alloc((size_t)N2 * 4);
  float*    u2a    = (float*)alloc((size_t)N2 * 4);
  float*    u2b    = (float*)alloc((size_t)N2 * 4);
  _Float16* h2p    = (_Float16*)alloc((size_t)M * 32 * 2);
  _Float16* fpw16  = (_Float16*)alloc((size_t)512 * 32 * 2);
  _Float16* spw1_16= (_Float16*)alloc((size_t)512 * 512 * 2);
  _Float16* spw2_16= (_Float16*)alloc((size_t)32 * 512 * 2);
  _Float16* spw3_16= (_Float16*)alloc((size_t)512 * 32 * 2);
  _Float16* big1h  = (_Float16*)alloc((size_t)M * 512 * 2);
  _Float16* big2h  = (_Float16*)alloc((size_t)M * 512 * 2);
  _Float16* h5h    = (_Float16*)alloc((size_t)M * 32 * 2);

  // --- dictionary prep ---
  k_normalize<<<1, 64, 0, stream>>>(W1, Wn1, 16, 25);
  k_normalize<<<1, 64, 0, stream>>>(W2, Wn2, 28, 400);
  k_gram<<<(81 * 32 * 32 + 255) / 256, 256, 0, stream>>>(Wn1, Gt1, 16, 1);
  k_gram<<<(81 * 32 * 32 + 255) / 256, 256, 0, stream>>>(Wn2, Gt2, 28, 16);

  // --- LCA layer 1 (16 features) ---
  k_drive<<<(N1 + 255) / 256, 256, 0, stream>>>(x, Wn1, drive1, 8, 1, 16);
  k_zero<<<(N1 + 255) / 256, 256, 0, stream>>>(u1a, N1);
  for (int it = 0; it < 500; ++it) {
    float* ui = (it & 1) ? u1b : u1a;
    float* uo = (it & 1) ? u1a : u1b;
    k_lca_step<16><<<32, 128, 0, stream>>>(ui, uo, drive1, Gt1);
  }
  k_thresh<<<(N1 + 255) / 256, 256, 0, stream>>>(u1a, h1, N1);

  // --- LCA layer 2 (28 features) ---
  k_drive<<<(N2 + 255) / 256, 256, 0, stream>>>(h1, Wn2, drive2, 8, 16, 28);
  k_zero<<<(N2 + 255) / 256, 256, 0, stream>>>(u2a, N2);
  for (int it = 0; it < 500; ++it) {
    float* ui = (it & 1) ? u2b : u2a;
    float* uo = (it & 1) ? u2a : u2b;
    k_lca_step<28><<<32, 128, 0, stream>>>(ui, uo, drive2, Gt2);
  }
  k_thresh_pad<<<(M * 32 + 255) / 256, 256, 0, stream>>>(u2a, h2p, M);

  // --- weight conversion to padded f16 ---
  k_cvt_pad<<<(512 * 32 + 255) / 256, 256, 0, stream>>>(fp_w, fpw16, 500, 28, 512, 32);
  k_cvt_pad<<<(512 * 512 + 255) / 256, 256, 0, stream>>>(sp_w1, spw1_16, 500, 500, 512, 512);
  k_cvt_pad<<<(32 * 512 + 255) / 256, 256, 0, stream>>>(sp_w2, spw2_16, 28, 500, 32, 512);
  k_cvt_pad<<<(512 * 32 + 255) / 256, 256, 0, stream>>>(sp_w3, spw3_16, 500, 28, 512, 32);

  // --- MLP chain (rows = b,c,y; contraction over last dim) ---
  {
    int t1 = (M / 16) * (512 / 16);
    int t2 = (M / 16) * (32 / 16);
    k_gemm16<1><<<(t1 + 3) / 4, 128, 0, stream>>>(h2p,   fpw16,   fp_b,  big1h, M, 32,  500, 512);
    k_gemm16<1><<<(t1 + 3) / 4, 128, 0, stream>>>(big1h, spw1_16, sp_b1, big2h, M, 512, 500, 512);
    k_gemm16<1><<<(t2 + 3) / 4, 128, 0, stream>>>(big2h, spw2_16, sp_b2, h5h,   M, 512, 28,  32);
    k_gemm16<0><<<(t1 + 3) / 4, 128, 0, stream>>>(h5h,   spw3_16, sp_b3, big1h, M, 32,  500, 512);
  }

  // --- final projection [8,392000] @ [10,392000]^T ---
  k_final<<<80, 256, 0, stream>>>(big1h, tp_w, tp_b, out);
}
